// MLPTorsoEquivariantLite_86887188398109
// MI455X (gfx1250) — matrix operand order repaired and checked by reading, not compiled, not run on hardware
//
#include <hip/hip_runtime.h>
#include <hip/hip_bf16.h>
#include <math.h>

// ---------------------------------------------------------------------------
// MLPTorsoEquivariantLite fused forward for gfx1250 (MI455X, wave32, WMMA).
// One block = 8 waves = 32 tokens. Phase A: per-token math, lane = feature,
// wave-uniform broadcasts via v_readlane. Phase B: z-head GEMMs on
// v_wmma_f32_16x16x32_bf16 with pre-permuted bf16 fragments (2x b128 per frag).
// ---------------------------------------------------------------------------

typedef __attribute__((ext_vector_type(16))) __bf16 v16bf;
typedef __attribute__((ext_vector_type(8)))  float  v8f;

union BF16x16 { v16bf v; unsigned short u[16]; };

__device__ __forceinline__ unsigned short f2bf(float f) {
  unsigned u = __float_as_uint(f);
  u += 0x7FFFu + ((u >> 16) & 1u);          // round-to-nearest-even
  return (unsigned short)(u >> 16);
}

// wave-uniform lane broadcast -> v_readlane_b32 (no LDS traffic)
__device__ __forceinline__ float bcast(float v, int l) {
  return __uint_as_float((unsigned)__builtin_amdgcn_readlane((int)__float_as_uint(v), l));
}

__device__ __forceinline__ float wsum(float v) {
  #pragma unroll
  for (int o = 16; o > 0; o >>= 1) v += __shfl_xor(v, o, 32);
  return v;
}

__device__ __forceinline__ float gelu(float x) {   // tanh approx (jax default)
  float t = 0.7978845608028654f * (x + 0.044715f * x * x * x);
  return 0.5f * x * (1.0f + tanhf(t));
}

__device__ __forceinline__ float ln32(float v, float s, float b) {
  float mu  = wsum(v) * (1.0f / 32.0f);
  float d   = v - mu;
  float var = wsum(d * d) * (1.0f / 32.0f);
  return d * rsqrtf(var + 1e-6f) * s + b;
}

__device__ __forceinline__ void msoftmax5(const float sc[5], const bool m[5], float w[5]) {
  float mx = -3.402823466e38f; bool any = false;
  #pragma unroll
  for (int s = 0; s < 5; s++) if (m[s]) { any = true; mx = fmaxf(mx, sc[s]); }
  float den = 0.f;
  #pragma unroll
  for (int s = 0; s < 5; s++) { float e = m[s] ? __expf(sc[s] - mx) : 0.f; w[s] = e; den += e; }
  float inv = (any && den > 0.f) ? 1.f / den : 0.f;
  #pragma unroll
  for (int s = 0; s < 5; s++) w[s] *= inv;
}

// Within a 32-wide K block, WMMA A-fragment element order: permutation that
// swaps octets 8..15 <-> 16..23 (lane-half / element-half striping).
__device__ __forceinline__ int perm32(int r) {
  return (((r >> 3) & 1) << 4) + (r & 7) + ((r & 16) >> 1);
}

struct Params {
  const float* x;
  const float *w_ego, *b_ego, *w_pair, *b_pair, *w_ter, *b_ter, *w_invtok, *b_invtok;
  const float *ln1s, *ln1b, *wq, *bq, *wk, *bk, *wv, *bv, *wo, *bo, *ln2s, *ln2b;
  const float *tw1, *tb1, *tw2, *tb2;
  const float *spls, *splb, *spw, *spb, *splt;
  const float *apwq, *apbq, *apwk, *apbk, *apwe, *apbe;
  const float *flatw, *flatb;
  const float *vrwego, *vrbego, *vrlns, *vrlnb, *vrwh, *vrbh, *vrwl, *vrbl;
  const float *zb1, *zl1s, *zl1b, *zb2, *zl2s, *zl2b;
  const unsigned short *pw1, *pw2;   // prepped bf16 B-matrices in fragment order
  float* out;
  int ntok;
};

#define TOKS 32

// Pre-convert an f32 [K x 128] weight matrix into bf16 B-fragment order:
// dst[kb][col][half][e] , k = kb*32 + perm32(half*16 + e), zero-padded to KP.
__global__ void prep_w_kernel(const float* __restrict__ w, unsigned short* __restrict__ dst,
                              int K, int total) {
  int idx = blockIdx.x * 256 + threadIdx.x;
  if (idx >= total) return;
  int e    = idx & 15;
  int half = (idx >> 4) & 1;
  int col  = (idx >> 5) & 127;
  int kb   = idx >> 12;
  int k = kb * 32 + perm32(half * 16 + e);
  float v = (k < K) ? w[k * 128 + col] : 0.f;
  dst[idx] = f2bf(v);
}

__global__ __launch_bounds__(256)
void mlp_torso_fused(Params p) {
  __shared__ unsigned short s_zpre[TOKS][352];   // bf16 z-concat, A-frag permuted
  __shared__ float          s_zbuf[TOKS][128];   // f32 inter-stage
  __shared__ unsigned short s_zbf [TOKS][128];   // bf16 LN1 out, A-frag permuted

  const int lane    = threadIdx.x & 31;
  const int wave    = threadIdx.x >> 5;
  const int tokBase = blockIdx.x * TOKS;
  const int f       = lane;
  const int fp      = perm32(f);                 // permuted position of feature f

  // ============================ Phase A ===================================
  for (int it = 0; it < 4; ++it) {
    const int lt  = wave * 4 + it;
    const int tok = tokBase + lt;
    if (tok >= p.ntok) {
      for (int k = lane; k < 352; k += 32) s_zpre[lt][k] = 0;
      continue;
    }
    const float* xr = p.x + (long)tok * 198;
    if (it < 3) __builtin_prefetch(p.x + (long)(tok + 1) * 198, 0, 0);

    // --- ego embed: 32 -> 64 (lane owns outputs f and f+32) ---
    float egof = xr[f];
    float aA = p.b_ego[f], aB = p.b_ego[f + 32];
    for (int k2 = 0; k2 < 32; k2++) {
      float e = bcast(egof, k2);
      aA += e * p.w_ego[k2 * 64 + f];
      aB += e * p.w_ego[k2 * 64 + f + 32];
    }
    float egoA = gelu(aA), egoB = gelu(aB);

    // --- pair embed: 10 -> 32 ---
    float pacc = p.b_pair[f];
    for (int k2 = 0; k2 < 10; k2++) pacc += xr[107 + k2] * p.w_pair[k2 * 32 + f];
    float pairE = gelu(pacc);

    // --- ter embed: 81 -> 64 ---
    float tAa = p.b_ter[f], tBa = p.b_ter[f + 32];
    for (int k2 = 0; k2 < 81; k2++) {
      float tv = xr[117 + k2];
      tAa += tv * p.w_ter[k2 * 64 + f];
      tBa += tv * p.w_ter[k2 * 64 + f + 32];
    }
    float terA = gelu(tAa), terB = gelu(tBa);

    // --- per-slot invariant tokens: inv(10) -> 32, masked ---
    bool  m[5]; float treg[5];
    #pragma unroll
    for (int s = 0; s < 5; s++) {
      const float* nb = xr + 32 + s * 15;
      float r0 = nb[0], r1 = nb[1], r2 = nb[2];
      bool  ms = (fabsf(r0) > 1e-6f) || (fabsf(r1) > 1e-6f) || (fabsf(r2) > 1e-6f);
      m[s] = ms;
      float a0 = nb[6], a1 = nb[7], a2 = nb[8];
      float an = sqrtf(a0 * a0 + a1 * a1 + a2 * a2);
      float inv[10] = { nb[9], nb[10], nb[11], nb[12], nb[13], nb[14], a0, a1, a2, an };
      float acc = p.b_invtok[f];
      #pragma unroll
      for (int k2 = 0; k2 < 10; k2++) acc += inv[k2] * p.w_invtok[k2 * 32 + f];
      treg[s] = ms ? gelu(acc) : 0.f;
    }

    // --- transformer block over 5 slots, dim 32 ---
    float yreg[5], yq[5], yk_[5], yv[5];
    #pragma unroll
    for (int s = 0; s < 5; s++) yreg[s] = ln32(treg[s], p.ln1s[f], p.ln1b[f]);
    #pragma unroll
    for (int s = 0; s < 5; s++) {
      float aq = p.bq[f], ak = p.bk[f], av = p.bv[f];
      for (int k2 = 0; k2 < 32; k2++) {
        float yv2 = bcast(yreg[s], k2);
        aq += yv2 * p.wq[k2 * 32 + f];
        ak += yv2 * p.wk[k2 * 32 + f];
        av += yv2 * p.wv[k2 * 32 + f];
      }
      yq[s] = aq; yk_[s] = ak; yv[s] = av;
    }
    // attention: 4 heads x 8 dims; head = lane>>3; reduce within 8-lane groups
    float yatt[5];
    #pragma unroll
    for (int qs = 0; qs < 5; qs++) {
      float sc[5];
      #pragma unroll
      for (int ks = 0; ks < 5; ks++) {
        float pp = yq[qs] * yk_[ks];
        pp += __shfl_xor(pp, 1, 32);
        pp += __shfl_xor(pp, 2, 32);
        pp += __shfl_xor(pp, 4, 32);
        sc[ks] = pp * 0.35355339059327373f;     // 1/sqrt(8)
      }
      float w5[5]; msoftmax5(sc, m, w5);
      float acc = 0.f;
      #pragma unroll
      for (int ks = 0; ks < 5; ks++) acc += w5[ks] * yv[ks];
      yatt[qs] = acc;
    }
    // output projection + masked residual
    #pragma unroll
    for (int s = 0; s < 5; s++) {
      float ao = p.bo[f];
      for (int k2 = 0; k2 < 32; k2++) ao += bcast(yatt[s], k2) * p.wo[k2 * 32 + f];
      treg[s] += m[s] ? ao : 0.f;
    }
    // FFN 32 -> 64 -> 32, masked residual
    #pragma unroll
    for (int s = 0; s < 5; s++) {
      float y2 = ln32(treg[s], p.ln2s[f], p.ln2b[f]);
      float hA = p.tb1[f], hB = p.tb1[f + 32];
      for (int k2 = 0; k2 < 32; k2++) {
        float yv2 = bcast(y2, k2);
        hA += yv2 * p.tw1[k2 * 64 + f];
        hB += yv2 * p.tw1[k2 * 64 + f + 32];
      }
      hA = gelu(hA); hB = gelu(hB);
      float ao = p.tb2[f];
      for (int k2 = 0; k2 < 32; k2++) ao += bcast(hA, k2) * p.tw2[k2 * 32 + f];
      for (int k2 = 0; k2 < 32; k2++) ao += bcast(hB, k2) * p.tw2[(k2 + 32) * 32 + f];
      treg[s] = m[s] ? (treg[s] + ao) : 0.f;
    }

    // --- pooling ---
    float invSum = 0.f, invMax = -1e9f; int cnt = 0; bool anyM = false;
    #pragma unroll
    for (int s = 0; s < 5; s++) {
      invSum += treg[s];
      if (m[s]) { cnt++; anyM = true; invMax = fmaxf(invMax, treg[s]); }
    }
    float invMean = invSum / (float)(cnt < 1 ? 1 : cnt);
    if (!anyM) invMax = 0.f;

    // --- sp soft pooling ---
    float scv[5];
    #pragma unroll
    for (int s = 0; s < 5; s++) {
      float yl = ln32(treg[s], p.spls[f], p.splb[f]);
      scv[s] = wsum(yl * p.spw[f]) + p.spb[0];
    }
    float temp = log1pf(__expf(p.splt[0])) + 0.001f;
    float tsc[5];
    #pragma unroll
    for (int s = 0; s < 5; s++) tsc[s] = temp * scv[s];
    float wts[5]; msoftmax5(tsc, m, wts);
    float invSoft = 0.f;
    #pragma unroll
    for (int s = 0; s < 5; s++) invSoft += wts[s] * treg[s];

    // --- additive attention pooling ---
    float qe = p.apbq[f];
    for (int k2 = 0; k2 < 32; k2++) qe += bcast(egoA, k2) * p.apwq[k2 * 32 + f];
    for (int k2 = 0; k2 < 32; k2++) qe += bcast(egoB, k2) * p.apwq[(k2 + 32) * 32 + f];
    float ev[5];
    #pragma unroll
    for (int s = 0; s < 5; s++) {
      float ke = p.apbk[f];
      for (int k2 = 0; k2 < 32; k2++) ke += bcast(treg[s], k2) * p.apwk[k2 * 32 + f];
      ev[s] = wsum(tanhf(qe + ke) * p.apwe[f]) + p.apbe[0];
    }
    float alp[5]; msoftmax5(ev, m, alp);
    float invAdd = 0.f;
    #pragma unroll
    for (int s = 0; s < 5; s++) invAdd += alp[s] * treg[s];

    // --- flat branch: 160 -> 32 ---
    float fl = p.flatb[f];
    #pragma unroll
    for (int s = 0; s < 5; s++)
      for (int j = 0; j < 32; j++) fl += bcast(treg[s], j) * p.flatw[(s * 32 + j) * 32 + f];
    float invFlat = 0.25f * gelu(fl);

    // --- vector readout ---
    float egoH = p.vrbego[f];
    for (int k2 = 0; k2 < 32; k2++) egoH += bcast(egoA, k2) * p.vrwego[k2 * 32 + f];
    for (int k2 = 0; k2 < 32; k2++) egoH += bcast(egoB, k2) * p.vrwego[(k2 + 32) * 32 + f];
    float l0[5], l1[5];
    #pragma unroll
    for (int s = 0; s < 5; s++) {
      float lv = ln32(treg[s], p.vrlns[f], p.vrlnb[f]);
      float h = p.vrbh[f];
      for (int k2 = 0; k2 < 32; k2++) h += bcast(lv,   k2) * p.vrwh[k2 * 32 + f];
      for (int k2 = 0; k2 < 32; k2++) h += bcast(egoH, k2) * p.vrwh[(k2 + 32) * 32 + f];
      h = gelu(h);
      l0[s] = wsum(h * p.vrwl[f * 2 + 0]) + p.vrbl[0];
      l1[s] = wsum(h * p.vrwl[f * 2 + 1]) + p.vrbl[1];
    }
    float a2h0[5], a2h1[5];
    msoftmax5(l0, m, a2h0);
    msoftmax5(l1, m, a2h1);

    // --- emit 332-wide z-concat as bf16 into LDS (A-fragment permuted) ---
    unsigned short* zp = s_zpre[lt];
    zp[fp]        = f2bf(egoA);
    zp[32 + fp]   = f2bf(egoB);
    zp[64 + fp]   = f2bf(pairE);
    zp[96 + fp]   = f2bf(terA);
    zp[128 + fp]  = f2bf(terB);
    zp[160 + fp]  = f2bf(invMean);
    zp[192 + fp]  = f2bf(invMax);
    zp[224 + fp]  = f2bf(invSoft);
    zp[256 + fp]  = f2bf(invAdd);
    zp[288 + fp]  = f2bf(invFlat);
    if (f < 12) {                 // v_r (6) then v_u (6): order [h][d]
      int j  = (f < 6) ? f : f - 6;
      int h2 = j / 3, d = j % 3;
      int off = (f < 6) ? d : (3 + d);   // r or u component
      float acc = 0.f;
      #pragma unroll
      for (int s = 0; s < 5; s++) {
        float comp = xr[32 + s * 15 + off];
        float aa = (h2 == 0) ? a2h0[s] : a2h1[s];
        acc += aa * comp;
      }
      zp[320 + perm32(f)] = f2bf(acc);
    }
    if (f < 20) zp[320 + perm32(12 + f)] = 0;   // zero pad logical 332..351
  }

  __syncthreads();

  // ====================== Phase B: z GEMMs on WMMA ========================
  // GEMM1: [32 x 352(bf16)] x [332->352 x 128] ; wave = N-tile, 2 M-tiles.
  {
    const int col = wave * 16 + (lane & 15);
    const int hi  = (lane >= 16) ? 1 : 0;
    v8f c0 = {}; v8f c1 = {};
    for (int kb = 0; kb < 11; ++kb) {
      BF16x16 bf = *(const BF16x16*)(p.pw1 + (((kb * 128 + col) * 2 + hi) << 4));
      BF16x16 a0 = *(const BF16x16*)&s_zpre[(lane & 15)     ][kb * 32 + hi * 16];
      BF16x16 a1 = *(const BF16x16*)&s_zpre[(lane & 15) + 16][kb * 32 + hi * 16];
      c0 = __builtin_amdgcn_wmma_f32_16x16x32_bf16(false, a0.v, false, bf.v, (short)0, c0, false, false);
      c1 = __builtin_amdgcn_wmma_f32_16x16x32_bf16(false, a1.v, false, bf.v, (short)0, c1, false, false);
    }
    const float bb = p.zb1[col];
    #pragma unroll
    for (int r = 0; r < 8; r++) {
      int row = (lane < 16) ? r : r + 8;
      s_zbuf[row][col]      = gelu(c0[r] + bb);
      s_zbuf[row + 16][col] = gelu(c1[r] + bb);
    }
  }
  __syncthreads();

  // LayerNorm over 128 -> bf16 (A of GEMM2, permuted store)
  #pragma unroll
  for (int rr = 0; rr < 4; rr++) {
    int row = wave * 4 + rr;
    float v0 = s_zbuf[row][lane],      v1 = s_zbuf[row][lane + 32];
    float v2 = s_zbuf[row][lane + 64], v3 = s_zbuf[row][lane + 96];
    float mu = wsum(v0 + v1 + v2 + v3) * (1.f / 128.f);
    float d0 = v0 - mu, d1 = v1 - mu, d2 = v2 - mu, d3 = v3 - mu;
    float var = wsum(d0 * d0 + d1 * d1 + d2 * d2 + d3 * d3) * (1.f / 128.f);
    float rs = rsqrtf(var + 1e-6f);
    s_zbf[row][fp]      = f2bf(d0 * rs * p.zl1s[lane]      + p.zl1b[lane]);
    s_zbf[row][32 + fp] = f2bf(d1 * rs * p.zl1s[lane + 32] + p.zl1b[lane + 32]);
    s_zbf[row][64 + fp] = f2bf(d2 * rs * p.zl1s[lane + 64] + p.zl1b[lane + 64]);
    s_zbf[row][96 + fp] = f2bf(d3 * rs * p.zl1s[lane + 96] + p.zl1b[lane + 96]);
  }
  __syncthreads();

  // GEMM2: [32 x 128(bf16)] x [128 x 128]
  {
    const int col = wave * 16 + (lane & 15);
    const int hi  = (lane >= 16) ? 1 : 0;
    v8f c0 = {}; v8f c1 = {};
    #pragma unroll
    for (int kb = 0; kb < 4; ++kb) {
      BF16x16 bf = *(const BF16x16*)(p.pw2 + (((kb * 128 + col) * 2 + hi) << 4));
      BF16x16 a0 = *(const BF16x16*)&s_zbf[(lane & 15)     ][kb * 32 + hi * 16];
      BF16x16 a1 = *(const BF16x16*)&s_zbf[(lane & 15) + 16][kb * 32 + hi * 16];
      c0 = __builtin_amdgcn_wmma_f32_16x16x32_bf16(false, a0.v, false, bf.v, (short)0, c0, false, false);
      c1 = __builtin_amdgcn_wmma_f32_16x16x32_bf16(false, a1.v, false, bf.v, (short)0, c1, false, false);
    }
    const float bb = p.zb2[col];
    #pragma unroll
    for (int r = 0; r < 8; r++) {
      int row = (lane < 16) ? r : r + 8;
      s_zbuf[row][col]      = gelu(c0[r] + bb);
      s_zbuf[row + 16][col] = gelu(c1[r] + bb);
    }
  }
  __syncthreads();

  // Final LayerNorm over 128 -> global output (f32)
  #pragma unroll
  for (int rr = 0; rr < 4; rr++) {
    int row = wave * 4 + rr;
    int tok = tokBase + row;
    if (tok >= p.ntok) continue;
    float v0 = s_zbuf[row][lane],      v1 = s_zbuf[row][lane + 32];
    float v2 = s_zbuf[row][lane + 64], v3 = s_zbuf[row][lane + 96];
    float mu = wsum(v0 + v1 + v2 + v3) * (1.f / 128.f);
    float d0 = v0 - mu, d1 = v1 - mu, d2 = v2 - mu, d3 = v3 - mu;
    float var = wsum(d0 * d0 + d1 * d1 + d2 * d2 + d3 * d3) * (1.f / 128.f);
    float rs = rsqrtf(var + 1e-6f);
    float* o = p.out + (long)tok * 128;
    o[lane]      = d0 * rs * p.zl2s[lane]      + p.zl2b[lane];
    o[lane + 32] = d1 * rs * p.zl2s[lane + 32] + p.zl2b[lane + 32];
    o[lane + 64] = d2 * rs * p.zl2s[lane + 64] + p.zl2b[lane + 64];
    o[lane + 96] = d3 * rs * p.zl2s[lane + 96] + p.zl2b[lane + 96];
  }
}

extern "C" void kernel_launch(void* const* d_in, const int* in_sizes, int n_in,
                              void* d_out, int out_size, void* d_ws, size_t ws_size,
                              hipStream_t stream) {
  (void)n_in; (void)ws_size; (void)out_size;
  unsigned short* pw1 = (unsigned short*)d_ws;            // 352*128 bf16
  unsigned short* pw2 = pw1 + 352 * 128;                  // 128*128 bf16

  // Prep: bf16 B-matrices in WMMA fragment order (zero-padded K).
  const float* zw1 = (const float*)d_in[46];
  const float* zw2 = (const float*)d_in[50];
  {
    int tot1 = (352 / 32) * 128 * 32;   // 45056
    int tot2 = (128 / 32) * 128 * 32;   // 16384
    prep_w_kernel<<<(tot1 + 255) / 256, 256, 0, stream>>>(zw1, pw1, 332, tot1);
    prep_w_kernel<<<(tot2 + 255) / 256, 256, 0, stream>>>(zw2, pw2, 128, tot2);
  }

  Params p;
  p.x        = (const float*)d_in[0];
  p.w_ego    = (const float*)d_in[1];  p.b_ego    = (const float*)d_in[2];
  p.w_pair   = (const float*)d_in[3];  p.b_pair   = (const float*)d_in[4];
  p.w_ter    = (const float*)d_in[5];  p.b_ter    = (const float*)d_in[6];
  p.w_invtok = (const float*)d_in[7];  p.b_invtok = (const float*)d_in[8];
  p.ln1s     = (const float*)d_in[9];  p.ln1b     = (const float*)d_in[10];
  p.wq       = (const float*)d_in[11]; p.bq       = (const float*)d_in[12];
  p.wk       = (const float*)d_in[13]; p.bk       = (const float*)d_in[14];
  p.wv       = (const float*)d_in[15]; p.bv       = (const float*)d_in[16];
  p.wo       = (const float*)d_in[17]; p.bo       = (const float*)d_in[18];
  p.ln2s     = (const float*)d_in[19]; p.ln2b     = (const float*)d_in[20];
  p.tw1      = (const float*)d_in[21]; p.tb1      = (const float*)d_in[22];
  p.tw2      = (const float*)d_in[23]; p.tb2      = (const float*)d_in[24];
  p.spls     = (const float*)d_in[25]; p.splb     = (const float*)d_in[26];
  p.spw      = (const float*)d_in[27]; p.spb      = (const float*)d_in[28];
  p.splt     = (const float*)d_in[29];
  p.apwq     = (const float*)d_in[30]; p.apbq     = (const float*)d_in[31];
  p.apwk     = (const float*)d_in[32]; p.apbk     = (const float*)d_in[33];
  p.apwe     = (const float*)d_in[34]; p.apbe     = (const float*)d_in[35];
  p.flatw    = (const float*)d_in[36]; p.flatb    = (const float*)d_in[37];
  p.vrwego   = (const float*)d_in[38]; p.vrbego   = (const float*)d_in[39];
  p.vrlns    = (const float*)d_in[40]; p.vrlnb    = (const float*)d_in[41];
  p.vrwh     = (const float*)d_in[42]; p.vrbh     = (const float*)d_in[43];
  p.vrwl     = (const float*)d_in[44]; p.vrbl     = (const float*)d_in[45];
  p.zb1      = (const float*)d_in[47];
  p.zl1s     = (const float*)d_in[48]; p.zl1b     = (const float*)d_in[49];
  p.zb2      = (const float*)d_in[51];
  p.zl2s     = (const float*)d_in[52]; p.zl2b     = (const float*)d_in[53];
  p.pw1      = pw1;
  p.pw2      = pw2;
  p.out      = (float*)d_out;
  p.ntok     = in_sizes[0] / 198;

  int blocks = (p.ntok + TOKS - 1) / TOKS;
  mlp_torso_fused<<<blocks, 256, 0, stream>>>(p);
}
